// SEATBlock_45578192945778
// MI455X (gfx1250) — matrix operand order repaired and checked by reading, not compiled, not run on hardware
//
#include <hip/hip_runtime.h>
#include <math.h>

// ---------------------------------------------------------------------------
// Types / WMMA helpers (CDNA5 gfx1250, wave32)
// ---------------------------------------------------------------------------
typedef __attribute__((ext_vector_type(16))) _Float16 v16h;
typedef __attribute__((ext_vector_type(8)))  float    v8f;

__device__ __forceinline__ v8f zero8() {
    v8f z = {0.f, 0.f, 0.f, 0.f, 0.f, 0.f, 0.f, 0.f};
    return z;
}

__device__ __forceinline__ v8f wmma_f16(v16h a, v16h b, v8f c) {
    // (neg_a, A, neg_b, B, c_mod, C, reuse_a, reuse_b)
    return __builtin_amdgcn_wmma_f32_16x16x32_f16(false, a, false, b, (short)0, c,
                                                  false, false);
}

// Async global->LDS 16B copy (ASYNCcnt-tracked) + wait helpers.
__device__ __forceinline__ void async_cp16(const void* gptr, void* lptr) {
    unsigned lds = (unsigned)(uintptr_t)lptr;  // LDS addr = low 32 bits of generic
    asm volatile("global_load_async_to_lds_b128 %0, %1, off"
                 :: "v"(lds), "v"(gptr) : "memory");
}
__device__ __forceinline__ void async_wait0() {
    asm volatile("s_wait_asynccnt 0x0" ::: "memory");
}

// A fragment (16x32 f16). Per ISA 7.12.2: lanes 0-15 -> M=0..15, halves 0..7 =
// K(k0+koff .. +7), halves 8..15 = K(k0+koff+16 ..); koff = 0 (lanes<16) / 8.
__device__ __forceinline__ v16h ldfragA(const _Float16* tile, int k0, int ldm) {
    int lane = threadIdx.x & 31;
    const _Float16* p = tile + (lane & 15) * ldm + k0 + ((lane & 16) ? 8 : 0);
    v16h r;
    ((float4*)&r)[0] = *(const float4*)p;
    ((float4*)&r)[1] = *(const float4*)(p + 16);
    return r;
}

// B fragment (32x16 f16), treated as rows of a [N][K] tile. Same K order as A:
// halves 0..7 = K(k0+koff .. +7), halves 8..15 = K(k0+koff+16 ..); koff = 0 (lanes<16) / 8.
__device__ __forceinline__ v16h ldfragB(const _Float16* tile, int k0, int ldm) {
    int lane = threadIdx.x & 31;
    const _Float16* p = tile + (lane & 15) * ldm + k0 + ((lane & 16) ? 8 : 0);
    v16h r;
    ((float4*)&r)[0] = *(const float4*)p;
    ((float4*)&r)[1] = *(const float4*)(p + 16);
    return r;
}

__device__ __forceinline__ float gelu_exact(float x) {
    return 0.5f * x * (1.f + erff(x * 0.7071067811865476f));
}

// window index helper: wi in [0,1024), tok in [0,64) -> flat token index
__device__ __forceinline__ int win_tok_to_g(int wi, int tok) {
    int b = wi >> 6, w6 = wi & 63;
    int hw = w6 >> 3, ww = w6 & 7;
    int r = tok >> 3, cc = tok & 7;
    return b * 4096 + (hw * 8 + r) * 64 + (ww * 8 + cc);
}

// ---------------------------------------------------------------------------
// Weight convert: (K x N f32 row-major) -> f16 [N][K] (column-major for B tiles)
// ---------------------------------------------------------------------------
__global__ __launch_bounds__(256) void seat_wt_kernel(const float* __restrict__ in,
                                                      _Float16* __restrict__ out,
                                                      int K, int N) {
    int i = blockIdx.x * 256 + threadIdx.x;
    if (i < K * N) {
        int k = i / N, n = i - k * N;
        out[(size_t)n * K + k] = (_Float16)in[i];
    }
}

// ---------------------------------------------------------------------------
// LayerNorm over C=256, wave-per-token (8 tokens / 256-thread block)
// ---------------------------------------------------------------------------
template <bool F32OUT>
__global__ __launch_bounds__(256) void seat_ln_kernel(const float* __restrict__ in,
                                                      const float* __restrict__ w,
                                                      const float* __restrict__ b,
                                                      _Float16* __restrict__ outh,
                                                      float* __restrict__ outf) {
    int wave = threadIdx.x >> 5, lane = threadIdx.x & 31;
    size_t g = (size_t)blockIdx.x * 8 + wave;
    const float* row = in + g * 256;
    float v[8];
    float s = 0.f;
#pragma unroll
    for (int i = 0; i < 8; i++) { v[i] = row[lane + i * 32]; s += v[i]; }
#pragma unroll
    for (int off = 16; off; off >>= 1) s += __shfl_xor(s, off, 32);
    float mean = s * (1.f / 256.f);
    float q = 0.f;
#pragma unroll
    for (int i = 0; i < 8; i++) { float d = v[i] - mean; q += d * d; }
#pragma unroll
    for (int off = 16; off; off >>= 1) q += __shfl_xor(q, off, 32);
    float rstd = rsqrtf(q * (1.f / 256.f) + 1e-5f);
#pragma unroll
    for (int i = 0; i < 8; i++) {
        int c = lane + i * 32;
        float z = (v[i] - mean) * rstd * w[c] + b[c];
        outh[g * 256 + c] = (_Float16)z;
        if (F32OUT) outf[g * 256 + c] = z;
    }
}

// ---------------------------------------------------------------------------
// Generic f16 WMMA GEMM: C[M,N] = A[M,K] * Bcm[N,K]^T + bias (+ epilogue)
// block tile 64x32, 8 waves of one 16x16 WMMA tile each, K chunked by 32.
// Tiles staged with GLOBAL_LOAD_ASYNC_TO_LDS_B128 (ASYNCcnt).
// EPI 0: f32 out = acc+bias     (qkv)
// EPI 1: f32 out = acc+bias+add (proj+shortcut, fc2+x1)
// EPI 2: f16 out = gelu(acc+bias) (fc1)
// ---------------------------------------------------------------------------
template <int EPI>
__global__ __launch_bounds__(256) void seat_gemm_kernel(
    const _Float16* __restrict__ A, const _Float16* __restrict__ Bcm,
    const float* __restrict__ bias, const float* __restrict__ addend,
    float* __restrict__ outf, _Float16* __restrict__ outh, int K, int N) {
    __shared__ __align__(16) _Float16 As[64 * 40];
    __shared__ __align__(16) _Float16 Bs[32 * 40];
    int m0 = blockIdx.x * 64, n0 = blockIdx.y * 32;
    int t = threadIdx.x, wave = t >> 5, lane = t & 31;
    int tm = (wave >> 1) * 16, tn = (wave & 1) * 16;
    int ar = t >> 2, aq = (t & 3) * 8;
    v8f acc = zero8();
    for (int k0 = 0; k0 < K; k0 += 32) {
        async_cp16(&A[(size_t)(m0 + ar) * K + k0 + aq], &As[ar * 40 + aq]);
        if (t < 128)
            async_cp16(&Bcm[(size_t)(n0 + ar) * K + k0 + aq], &Bs[ar * 40 + aq]);
        if (k0 + 32 < K)
            __builtin_prefetch(&A[(size_t)(m0 + ar) * K + k0 + 32 + aq], 0, 0);
        async_wait0();
        __syncthreads();
        v16h af = ldfragA(As + tm * 40, 0, 40);
        v16h bf = ldfragB(Bs + tn * 40, 0, 40);
        acc = wmma_f16(af, bf, acc);
        __syncthreads();
    }
    int mb = m0 + tm + ((lane & 16) ? 8 : 0);
    int n = n0 + tn + (lane & 15);
    float bn = bias[n];
#pragma unroll
    for (int vg = 0; vg < 8; vg++) {
        size_t o = (size_t)(mb + vg) * N + n;
        float r = acc[vg] + bn;
        if (EPI == 0) outf[o] = r;
        if (EPI == 1) outf[o] = r + addend[o];
        if (EPI == 2) outh[o] = (_Float16)gelu_exact(r);
    }
}

// ---------------------------------------------------------------------------
// Complex f16 WMMA GEMM: Z = A * W with A=(Ar,Ai)[M,K], W=(Br,Bi)[N][K] col-major
// Zr = ArBr - AiBi ; Zi = ArBi + AiBr.  GELU epilogue for layer 1 (f16 out).
// ---------------------------------------------------------------------------
template <bool GELU>
__global__ __launch_bounds__(256) void seat_cgemm_kernel(
    const _Float16* __restrict__ Ar, const _Float16* __restrict__ Ai,
    const _Float16* __restrict__ Br, const _Float16* __restrict__ Bi,
    float* __restrict__ outRf, float* __restrict__ outIf,
    _Float16* __restrict__ outRh, _Float16* __restrict__ outIh, int K, int N) {
    __shared__ __align__(16) _Float16 Ars[64 * 40];
    __shared__ __align__(16) _Float16 Ais[64 * 40];
    __shared__ __align__(16) _Float16 Brs[32 * 40];
    __shared__ __align__(16) _Float16 Bis[32 * 40];
    int m0 = blockIdx.x * 64, n0 = blockIdx.y * 32;
    int t = threadIdx.x, wave = t >> 5, lane = t & 31;
    int tm = (wave >> 1) * 16, tn = (wave & 1) * 16;
    int ar = t >> 2, aq = (t & 3) * 8;
    v8f accP = zero8(), accQ = zero8(), accI = zero8();
    for (int k0 = 0; k0 < K; k0 += 32) {
        async_cp16(&Ar[(size_t)(m0 + ar) * K + k0 + aq], &Ars[ar * 40 + aq]);
        async_cp16(&Ai[(size_t)(m0 + ar) * K + k0 + aq], &Ais[ar * 40 + aq]);
        if (t < 128) {
            async_cp16(&Br[(size_t)(n0 + ar) * K + k0 + aq], &Brs[ar * 40 + aq]);
            async_cp16(&Bi[(size_t)(n0 + ar) * K + k0 + aq], &Bis[ar * 40 + aq]);
        }
        if (k0 + 32 < K)
            __builtin_prefetch(&Ar[(size_t)(m0 + ar) * K + k0 + 32 + aq], 0, 0);
        async_wait0();
        __syncthreads();
        v16h fAr = ldfragA(Ars + tm * 40, 0, 40);
        v16h fAi = ldfragA(Ais + tm * 40, 0, 40);
        v16h fBr = ldfragB(Brs + tn * 40, 0, 40);
        v16h fBi = ldfragB(Bis + tn * 40, 0, 40);
        accP = wmma_f16(fAr, fBr, accP);
        accQ = wmma_f16(fAi, fBi, accQ);
        accI = wmma_f16(fAr, fBi, accI);
        accI = wmma_f16(fAi, fBr, accI);
        __syncthreads();
    }
    int mb = m0 + tm + ((lane & 16) ? 8 : 0);
    int n = n0 + tn + (lane & 15);
#pragma unroll
    for (int vg = 0; vg < 8; vg++) {
        size_t o = (size_t)(mb + vg) * N + n;
        float R = accP[vg] - accQ[vg];
        float I = accI[vg];
        if (GELU) {
            outRh[o] = (_Float16)gelu_exact(R);
            outIh[o] = (_Float16)gelu_exact(I);
        } else {
            outRf[o] = R;
            outIf[o] = I;
        }
    }
}

// ---------------------------------------------------------------------------
// Agent-attention per (window, head): all matmuls on WMMA from LDS tiles.
// ---------------------------------------------------------------------------
__device__ __forceinline__ void softmax64(int t, float* Sf, _Float16* P) {
    if (t < 64) {
        float mx = -1e30f;
        for (int n = 0; n < 64; n++) mx = fmaxf(mx, Sf[t * 64 + n]);
        float sum = 0.f;
        for (int n = 0; n < 64; n++) {
            float e = expf(Sf[t * 64 + n] - mx);
            Sf[t * 64 + n] = e;
            sum += e;
        }
        float inv = 1.f / sum;
        for (int n = 0; n < 64; n++) P[t * 72 + n] = (_Float16)(Sf[t * 64 + n] * inv);
    }
}

__global__ __launch_bounds__(256) void seat_attn_kernel(
    const float* __restrict__ qkv, const float* __restrict__ an_bias,
    const float* __restrict__ na_bias, const float* __restrict__ ah_bias,
    const float* __restrict__ aw_bias, const float* __restrict__ ha_bias,
    const float* __restrict__ wa_bias, float* __restrict__ xa) {
    __shared__ __align__(16) _Float16 qs[64 * 40];
    __shared__ __align__(16) _Float16 ks[64 * 40];
    __shared__ __align__(16) _Float16 vT[32 * 72];
    __shared__ __align__(16) _Float16 P[64 * 72];
    __shared__ __align__(16) _Float16 avT[32 * 72];
    __shared__ float Sf[64 * 64];
    int wi = blockIdx.x, h = blockIdx.y;
    int t = threadIdx.x, wave = t >> 5, lane = t & 31;
    const float scale = 0.17677669529663687f;  // 32^-0.5

    // stage q, k (token-major, unscaled) and v (transposed: [ch][tok]) as f16
    for (int i = t; i < 64 * 32; i += 256) {
        int tok = i >> 5, c = i & 31;
        size_t base = (size_t)win_tok_to_g(wi, tok) * 768 + h * 32 + c;
        qs[tok * 40 + c] = (_Float16)qkv[base];
        ks[tok * 40 + c] = (_Float16)qkv[base + 256];
        vT[c * 72 + tok] = (_Float16)qkv[base + 512];
    }
    __syncthreads();

    // S = scale*(q k^T) + pb   (16 tiles, 2 per wave)
#pragma unroll
    for (int it = 0; it < 2; it++) {
        int tile = wave * 2 + it, ti = tile >> 2, tj = tile & 3;
        v16h a = ldfragA(qs + ti * 16 * 40, 0, 40);
        v16h b = ldfragB(ks + tj * 16 * 40, 0, 40);
        v8f c = zero8();
        c = wmma_f16(a, b, c);
        int mb = ti * 16 + ((lane & 16) ? 8 : 0), n = tj * 16 + (lane & 15);
#pragma unroll
        for (int vg = 0; vg < 8; vg++) {
            int aa = mb + vg;
            float pb = an_bias[(h * 64 + aa) * 64 + n] +
                       ah_bias[(h * 64 + aa) * 8 + (n >> 3)] +
                       aw_bias[(h * 64 + aa) * 8 + (n & 7)];
            Sf[aa * 64 + n] = scale * c[vg] + pb;
        }
    }
    __syncthreads();
    softmax64(t, Sf, P);
    __syncthreads();

    // agent_v = P @ v  -> avT[ch][a] (f16)
    {
        int ti = wave >> 1, tj = wave & 1;
        v8f c = zero8();
#pragma unroll
        for (int k0 = 0; k0 < 64; k0 += 32) {
            v16h a = ldfragA(P + ti * 16 * 72, k0, 72);
            v16h b = ldfragB(vT + tj * 16 * 72, k0, 72);
            c = wmma_f16(a, b, c);
        }
        int mb = ti * 16 + ((lane & 16) ? 8 : 0), ch = tj * 16 + (lane & 15);
#pragma unroll
        for (int vg = 0; vg < 8; vg++) avT[ch * 72 + mb + vg] = (_Float16)c[vg];
    }

    // T = scale*(q q^T) + ab   (q_attn logits; rows = query token, cols = agent)
#pragma unroll
    for (int it = 0; it < 2; it++) {
        int tile = wave * 2 + it, ti = tile >> 2, tj = tile & 3;
        v16h a = ldfragA(qs + ti * 16 * 40, 0, 40);
        v16h b = ldfragB(qs + tj * 16 * 40, 0, 40);
        v8f c = zero8();
        c = wmma_f16(a, b, c);
        int mb = ti * 16 + ((lane & 16) ? 8 : 0), acol = tj * 16 + (lane & 15);
#pragma unroll
        for (int vg = 0; vg < 8; vg++) {
            int n = mb + vg;
            float ab = na_bias[(h * 64 + acol) * 64 + n] +
                       ha_bias[(h * 8 + (n >> 3)) * 64 + acol] +
                       wa_bias[(h * 8 + (n & 7)) * 64 + acol];
            Sf[n * 64 + acol] = scale * c[vg] + ab;
        }
    }
    __syncthreads();
    softmax64(t, Sf, P);
    __syncthreads();

    // xa = Pq @ agent_v
    {
        int ti = wave >> 1, tj = wave & 1;
        v8f c = zero8();
#pragma unroll
        for (int k0 = 0; k0 < 64; k0 += 32) {
            v16h a = ldfragA(P + ti * 16 * 72, k0, 72);
            v16h b = ldfragB(avT + tj * 16 * 72, k0, 72);
            c = wmma_f16(a, b, c);
        }
        int mb = ti * 16 + ((lane & 16) ? 8 : 0), ch = tj * 16 + (lane & 15);
#pragma unroll
        for (int vg = 0; vg < 8; vg++) {
            int tok = mb + vg;
            xa[(size_t)win_tok_to_g(wi, tok) * 256 + h * 32 + ch] = c[vg];
        }
    }
}

// ---------------------------------------------------------------------------
// Depthwise 3x3 conv (per-window, zero 'SAME' pad) on v; add to xa, emit f16.
// ---------------------------------------------------------------------------
__global__ __launch_bounds__(256) void seat_dwc_kernel(
    const float* __restrict__ qkv, const float* __restrict__ dwcw,
    const float* __restrict__ dwcb, const float* __restrict__ xa,
    _Float16* __restrict__ attnh) {
    int wi = blockIdx.x, c = threadIdx.x;
    float w9[9];
#pragma unroll
    for (int i = 0; i < 9; i++) w9[i] = dwcw[c * 9 + i];
    float bc = dwcb[c];
    for (int tok = 0; tok < 64; tok++) {
        int r = tok >> 3, cc = tok & 7;
        float acc = bc;
#pragma unroll
        for (int dy = -1; dy <= 1; dy++) {
#pragma unroll
            for (int dx = -1; dx <= 1; dx++) {
                int rr = r + dy, c2 = cc + dx;
                if (rr >= 0 && rr < 8 && c2 >= 0 && c2 < 8) {
                    int g2 = win_tok_to_g(wi, rr * 8 + c2);
                    acc += qkv[(size_t)g2 * 768 + 512 + c] * w9[(dy + 1) * 3 + (dx + 1)];
                }
            }
        }
        int g = win_tok_to_g(wi, tok);
        attnh[(size_t)g * 256 + c] = (_Float16)(xa[(size_t)g * 256 + c] + acc);
    }
}

// ---------------------------------------------------------------------------
// FFT branch: 64-pt DFT stages with LDS twiddles.
// F1: rfft along W. block=(b,h,wf), thread=c.  y(b,h,w,c) -> Y1(b,c,h,wf)
// ---------------------------------------------------------------------------
__global__ __launch_bounds__(256) void seat_f1_kernel(const float* __restrict__ y,
                                                      float* __restrict__ Y1r,
                                                      float* __restrict__ Y1i) {
    __shared__ float cs[64], sn[64];
    int bi = blockIdx.x;
    int wf = bi % 33, hh = (bi / 33) & 63, b = bi / (33 * 64);
    int t = threadIdx.x;
    if (t < 64) {
        float ang = -6.283185307179586f * (float)(t * wf) / 64.f;
        cs[t] = cosf(ang);
        sn[t] = sinf(ang);
    }
    __syncthreads();
    int c = t;
    float aR = 0.f, aI = 0.f;
    const float* base = y + ((size_t)b * 4096 + hh * 64) * 256 + c;
    for (int w = 0; w < 64; w++) {
        float v = base[(size_t)w * 256];
        aR += v * cs[w];
        aI += v * sn[w];
    }
    size_t o = (((size_t)b * 256 + c) * 64 + hh) * 33 + wf;
    Y1r[o] = aR;
    Y1i[o] = aI;
}

// F2: fft along H + transpose to GEMM layout. block=(b,k). Y1 -> Yt(b,k,wf,c) f16
__global__ __launch_bounds__(256) void seat_f2_kernel(const float* __restrict__ Y1r,
                                                      const float* __restrict__ Y1i,
                                                      _Float16* __restrict__ Ytr,
                                                      _Float16* __restrict__ Yti) {
    __shared__ float tr[64], ti[64];
    int b = blockIdx.x >> 6, k = blockIdx.x & 63;
    int t = threadIdx.x;
    if (t < 64) {
        float ang = -6.283185307179586f * (float)(t * k) / 64.f;
        tr[t] = cosf(ang);
        ti[t] = sinf(ang);
    }
    __syncthreads();
    for (int idx = t; idx < 256 * 33; idx += 256) {
        int wf = idx % 33, c = idx / 33;
        float aR = 0.f, aI = 0.f;
        size_t base = (((size_t)b * 256 + c) * 64) * 33 + wf;
        for (int hh = 0; hh < 64; hh++) {
            float r = Y1r[base + (size_t)hh * 33];
            float ii = Y1i[base + (size_t)hh * 33];
            aR += r * tr[hh] - ii * ti[hh];
            aI += r * ti[hh] + ii * tr[hh];
        }
        size_t o = (((size_t)b * 64 + k) * 33 + wf) * 256 + c;
        Ytr[o] = (_Float16)aR;
        Yti[o] = (_Float16)aI;
    }
}

// I1: inverse fft along H (with 1/64). block=(b,h). Z2(b,k,wf,c) -> G(b,h,wf,c)
__global__ __launch_bounds__(256) void seat_i1_kernel(const float* __restrict__ Z2r,
                                                      const float* __restrict__ Z2i,
                                                      float* __restrict__ Gr,
                                                      float* __restrict__ Gi) {
    __shared__ float tr[64], ti[64];
    int b = blockIdx.x >> 6, hh = blockIdx.x & 63;
    int t = threadIdx.x;
    if (t < 64) {
        float ang = 6.283185307179586f * (float)(t * hh) / 64.f;
        tr[t] = cosf(ang) * (1.f / 64.f);
        ti[t] = sinf(ang) * (1.f / 64.f);
    }
    __syncthreads();
    for (int idx = t; idx < 33 * 256; idx += 256) {
        int c = idx & 255, wf = idx >> 8;
        float aR = 0.f, aI = 0.f;
        for (int k = 0; k < 64; k++) {
            size_t o = (((size_t)b * 64 + k) * 33 + wf) * 256 + c;
            float r = Z2r[o], ii = Z2i[o];
            aR += r * tr[k] - ii * ti[k];
            aI += r * ti[k] + ii * tr[k];
        }
        size_t o = (((size_t)b * 64 + hh) * 33 + wf) * 256 + c;
        Gr[o] = aR;
        Gi[o] = aI;
    }
}

// I2: irfft along W (Hermitian weights, 1/64) and accumulate into out.
__global__ __launch_bounds__(256) void seat_i2_kernel(const float* __restrict__ Gr,
                                                      const float* __restrict__ Gi,
                                                      float* __restrict__ out) {
    __shared__ float c64[64], s64[64];
    int b = blockIdx.x >> 6, hh = blockIdx.x & 63;
    int c = threadIdx.x;
    if (c < 64) {
        float ang = 6.283185307179586f * (float)c / 64.f;
        c64[c] = cosf(ang);
        s64[c] = sinf(ang);
    }
    __syncthreads();
    size_t gbase = (((size_t)b * 64 + hh) * 33) * 256 + c;
    for (int w = 0; w < 64; w++) {
        float acc = 0.f;
        for (int wf = 0; wf <= 32; wf++) {
            int j = (w * wf) & 63;
            float wgt = (wf == 0 || wf == 32) ? 1.f : 2.f;
            size_t o = gbase + (size_t)wf * 256;
            acc += wgt * (Gr[o] * c64[j] - Gi[o] * s64[j]);
        }
        size_t oo = ((size_t)b * 4096 + hh * 64 + w) * 256 + c;
        out[oo] += acc * (1.f / 64.f);
    }
}

// ---------------------------------------------------------------------------
// Host launcher
// ---------------------------------------------------------------------------
extern "C" void kernel_launch(void* const* d_in, const int* in_sizes, int n_in,
                              void* d_out, int out_size, void* d_ws, size_t ws_size,
                              hipStream_t stream) {
    const float* x     = (const float*)d_in[0];
    const float* n1w   = (const float*)d_in[1];
    const float* n1b   = (const float*)d_in[2];
    const float* qkvw  = (const float*)d_in[3];
    const float* qkvb  = (const float*)d_in[4];
    const float* an_b  = (const float*)d_in[5];
    const float* na_b  = (const float*)d_in[6];
    const float* ah_b  = (const float*)d_in[7];
    const float* aw_b  = (const float*)d_in[8];
    const float* ha_b  = (const float*)d_in[9];
    const float* wa_b  = (const float*)d_in[10];
    const float* dwcw  = (const float*)d_in[11];
    const float* dwcb  = (const float*)d_in[12];
    const float* projw = (const float*)d_in[13];
    const float* projb = (const float*)d_in[14];
    const float* n2w   = (const float*)d_in[15];
    const float* n2b   = (const float*)d_in[16];
    const float* fc1w  = (const float*)d_in[17];
    const float* fc1b  = (const float*)d_in[18];
    const float* fc2w  = (const float*)d_in[19];
    const float* fc2b  = (const float*)d_in[20];
    const float* w1r   = (const float*)d_in[21];
    const float* w1i   = (const float*)d_in[22];
    const float* w2r   = (const float*)d_in[23];
    const float* w2i   = (const float*)d_in[24];
    float* out = (float*)d_out;

    const size_t M = 65536;             // B * H * W tokens
    const size_t FE = (size_t)16 * 64 * 33 * 256;  // freq-domain elements

    char* ws = (char*)d_ws;
    size_t off = 0;
    auto alloc = [&](size_t bytes) -> char* {
        char* p = ws + off;
        off += (bytes + 255) & ~(size_t)255;
        return p;
    };

    _Float16* ln1h  = (_Float16*)alloc(M * 256 * 2);   // reused as attnh
    _Float16* attnh = ln1h;
    float*    qkvf  = (float*)alloc(M * 768 * 4);      // reused as hidden (f16)
    _Float16* hidh  = (_Float16*)qkvf;
    float*    xaf   = (float*)alloc(M * 256 * 4);      // reused as y (f32)
    float*    yf    = xaf;
    float*    x1    = (float*)alloc(M * 256 * 4);
    _Float16* yh    = (_Float16*)alloc(M * 256 * 2);
    float*    Y1r   = (float*)alloc(FE * 4);           // reused as Z2r
    float*    Y1i   = (float*)alloc(FE * 4);           // reused as Z2i
    float*    Z2r   = Y1r;
    float*    Z2i   = Y1i;
    _Float16* Ytr   = (_Float16*)alloc(FE * 2);
    _Float16* Yti   = (_Float16*)alloc(FE * 2);
    _Float16* Z1r   = (_Float16*)alloc(FE * 2);
    _Float16* Z1i   = (_Float16*)alloc(FE * 2);
    float*    Gr    = (float*)alloc(FE * 4);
    float*    Gi    = (float*)alloc(FE * 4);
    _Float16* qkvT  = (_Float16*)alloc(768 * 256 * 2);
    _Float16* projT = (_Float16*)alloc(256 * 256 * 2);
    _Float16* fc1T  = (_Float16*)alloc((size_t)1024 * 256 * 2);
    _Float16* fc2T  = (_Float16*)alloc((size_t)256 * 1024 * 2);
    _Float16* w1rT  = (_Float16*)alloc(256 * 256 * 2);
    _Float16* w1iT  = (_Float16*)alloc(256 * 256 * 2);
    _Float16* w2rT  = (_Float16*)alloc(256 * 256 * 2);
    _Float16* w2iT  = (_Float16*)alloc(256 * 256 * 2);

    dim3 blk(256);

    // weight conversions (f32 [K,N] -> f16 [N][K])
    seat_wt_kernel<<<768, blk, 0, stream>>>(qkvw, qkvT, 256, 768);
    seat_wt_kernel<<<256, blk, 0, stream>>>(projw, projT, 256, 256);
    seat_wt_kernel<<<1024, blk, 0, stream>>>(fc1w, fc1T, 256, 1024);
    seat_wt_kernel<<<1024, blk, 0, stream>>>(fc2w, fc2T, 1024, 256);
    seat_wt_kernel<<<256, blk, 0, stream>>>(w1r, w1rT, 256, 256);
    seat_wt_kernel<<<256, blk, 0, stream>>>(w1i, w1iT, 256, 256);
    seat_wt_kernel<<<256, blk, 0, stream>>>(w2r, w2rT, 256, 256);
    seat_wt_kernel<<<256, blk, 0, stream>>>(w2i, w2iT, 256, 256);

    // LN1 -> f16
    seat_ln_kernel<false><<<8192, blk, 0, stream>>>(x, n1w, n1b, ln1h, nullptr);
    // QKV GEMM: (65536x256) @ (256x768)
    seat_gemm_kernel<0><<<dim3(1024, 24), blk, 0, stream>>>(ln1h, qkvT, qkvb, nullptr,
                                                            qkvf, nullptr, 256, 768);
    // agent attention per (window, head)
    seat_attn_kernel<<<dim3(1024, 8), blk, 0, stream>>>(qkvf, an_b, na_b, ah_b, aw_b,
                                                        ha_b, wa_b, xaf);
    // depthwise conv on v, add to xa -> f16
    seat_dwc_kernel<<<1024, blk, 0, stream>>>(qkvf, dwcw, dwcb, xaf, attnh);
    // projection + shortcut -> x1
    seat_gemm_kernel<1><<<dim3(1024, 8), blk, 0, stream>>>(attnh, projT, projb, x, x1,
                                                           nullptr, 256, 256);
    // LN2 -> y (f32 for FFT branch) + f16 (for MLP)
    seat_ln_kernel<true><<<8192, blk, 0, stream>>>(x1, n2w, n2b, yh, yf);
    // MLP fc1 + GELU -> hidden f16
    seat_gemm_kernel<2><<<dim3(1024, 32), blk, 0, stream>>>(yh, fc1T, fc1b, nullptr,
                                                            nullptr, hidh, 256, 1024);
    // MLP fc2 + x1 -> out
    seat_gemm_kernel<1><<<dim3(1024, 8), blk, 0, stream>>>(hidh, fc2T, fc2b, x1, out,
                                                           nullptr, 1024, 256);
    // spectral branch
    seat_f1_kernel<<<16 * 64 * 33, blk, 0, stream>>>(yf, Y1r, Y1i);
    seat_f2_kernel<<<1024, blk, 0, stream>>>(Y1r, Y1i, Ytr, Yti);
    seat_cgemm_kernel<true><<<dim3(528, 8), blk, 0, stream>>>(Ytr, Yti, w1rT, w1iT,
                                                              nullptr, nullptr, Z1r, Z1i,
                                                              256, 256);
    seat_cgemm_kernel<false><<<dim3(528, 8), blk, 0, stream>>>(Z1r, Z1i, w2rT, w2iT,
                                                               Z2r, Z2i, nullptr, nullptr,
                                                               256, 256);
    seat_i1_kernel<<<1024, blk, 0, stream>>>(Z2r, Z2i, Gr, Gi);
    seat_i2_kernel<<<1024, blk, 0, stream>>>(Gr, Gi, out);
}
